// lstm_classifier_79070347919820
// MI455X (gfx1250) — matrix operand order repaired and checked by reading, not compiled, not run on hardware
//
#include <hip/hip_runtime.h>

// ---------------------------------------------------------------------------
// Bi-LSTM text classifier for MI455X (gfx1250), bf16 WMMA + f32 accumulation.
// B=128, T=512, E=512, H=512, V=32000, C=10, PAD=1.
//
// Main kernel: 16 persistent workgroups (2 dirs x 8 batch tiles of 16 rows),
// 512 threads = 16 wave32s. Each wave owns 32 hidden columns (8 WMMA N-tiles:
// i,f,g,o gates x 2 halves). Per step: 256 v_wmma_f32_16x16x32_bf16 per wave
// (input projection fused with recurrence), LDS-resident h (bf16) as the
// shared A-matrix, register-resident f32 h/c with variable-length masking.
// Token sequences preloaded to LDS; embedding tile double-buffered so the
// gather for step s+1 overlaps the matmul of step s. Two barriers per step.
// ---------------------------------------------------------------------------

typedef __attribute__((ext_vector_type(8)))  float  v8f;
typedef __attribute__((ext_vector_type(16))) __bf16 v16bf;

#define B_  128
#define T_  512
#define E_  512
#define H_  512
#define G_  2048   // 4*H
#define C_  10

__device__ __forceinline__ unsigned short f2bf(float f) {
  unsigned u = __float_as_uint(f);
  unsigned r = (u + 0x7FFFu + ((u >> 16) & 1u)) >> 16;   // round-to-nearest-even
  return (unsigned short)r;
}
__device__ __forceinline__ float sigm(float x) { return 1.f / (1.f + __expf(-x)); }
__device__ __forceinline__ float tanh_f(float x) {
  x = fminf(fmaxf(x, -15.f), 15.f);
  float e = __expf(2.f * x);
  return (e - 1.f) / (e + 1.f);
}

union FragAB { uint4 u[2]; v16bf v; };

// A-fragment (16x32 bf16) from a row-major [16][512] bf16 tile in LDS.
// Lane L (m = L&15, khalf = L>>4) holds K = khalf*8 + {0..7} and +16.
__device__ __forceinline__ v16bf load_a(const unsigned short* rowptr, int k0, int khalf) {
  const uint4* p = (const uint4*)(rowptr + k0 + khalf * 8);
  FragAB a;
  a.u[0] = p[0];     // K = k0 + khalf*8 + 0..7
  a.u[1] = p[2];     // K = k0 + 16 + khalf*8 + 0..7  (+16 elems = +2 uint4)
  return a.v;
}

// B-fragment (32x16 bf16) from bf16 [K=512][N=2048] row-major weights:
// lane L holds column n = n0 + (L&15), K = (L>>4)*8 + {0..7} and +16.
__device__ __forceinline__ v16bf load_b(const unsigned short* wT, int kb, int n0) {
  const int l = threadIdx.x & 31;
  const unsigned short* p = wT + (size_t)(kb + (l >> 4) * 8) * G_ + n0 + (l & 15);
  unsigned short e[16];
#pragma unroll
  for (int s = 0; s < 16; s++) e[s] = p[(size_t)(s + (s >= 8 ? 8 : 0)) * G_];
  FragAB b;
  __builtin_memcpy(&b, e, sizeof(e));
  return b.v;
}

// ---------------------------------------------------------------------------
// Prep: w [2048][512] f32  ->  out [512][2048] bf16 (transposed + converted)
// ---------------------------------------------------------------------------
__global__ void k_wT_bf16(const float* __restrict__ w, unsigned short* __restrict__ out) {
  int idx = blockIdx.x * blockDim.x + threadIdx.x;
  if (idx >= G_ * 512) return;
  int n = idx >> 9;          // row of w    (0..2047)
  int k = idx & 511;         // col of w    (0..511)
  out[(size_t)k * G_ + n] = f2bf(w[idx]);
}

__global__ void k_bias(const float* __restrict__ a, const float* __restrict__ b,
                       float* __restrict__ o) {
  int i = blockIdx.x * blockDim.x + threadIdx.x;
  if (i < G_) o[i] = a[i] + b[i];
}

__global__ void k_lengths(const int* __restrict__ text, int* __restrict__ lens) {
  __shared__ int cnt;
  int b = blockIdx.x, t = threadIdx.x;
  if (t == 0) cnt = 0;
  __syncthreads();
  unsigned long long m = __ballot(text[b * T_ + t] != 1);
  if ((t & 31) == 0) atomicAdd(&cnt, (int)__popcll(m));
  __syncthreads();
  if (t == 0) lens[b] = cnt;
}

// ---------------------------------------------------------------------------
// Main fused bi-LSTM. grid = 16, block = 512.
// ---------------------------------------------------------------------------
__global__ __launch_bounds__(512, 1)
void k_bilstm(const int* __restrict__ text, const float* __restrict__ emb,
              const unsigned short* __restrict__ wT_ih_f,
              const unsigned short* __restrict__ wT_hh_f,
              const unsigned short* __restrict__ wT_ih_r,
              const unsigned short* __restrict__ wT_hh_r,
              const float* __restrict__ bias_f, const float* __restrict__ bias_r,
              const int* __restrict__ lens, float* __restrict__ hidden) {
  __shared__ unsigned short x_lds[2][16][E_];  // double-buffered bf16 x tile, 2x16KB
  __shared__ unsigned short h_lds[16][H_];     // bf16 hidden state, 16KB
  __shared__ int tok_lds[16][T_];              // per-row token sequence, 32KB
  __shared__ int len_s[16];

  const int dir = blockIdx.x & 1;
  const int b0  = (blockIdx.x >> 1) * 16;
  const unsigned short* wih  = dir ? wT_ih_r : wT_ih_f;
  const unsigned short* whh  = dir ? wT_hh_r : wT_hh_f;
  const float*          bias = dir ? bias_r  : bias_f;

  const int tid   = threadIdx.x;
  const int lane  = tid & 31;
  const int wave  = tid >> 5;
  const int nlane = lane & 15;
  const int khalf = lane >> 4;
  const int m     = lane & 15;          // A-row / C-col

  // ---- one-time LDS init: zero h, preload tokens + lengths ----
  for (int i = tid; i < 16 * H_; i += 512) ((unsigned short*)h_lds)[i] = 0;
  for (int i = tid; i < 16 * T_; i += 512) {
    int row = i >> 9, t = i & 511;
    tok_lds[row][t] = text[(b0 + row) * T_ + t];
  }
  if (tid < 16) len_s[tid] = lens[b0 + tid];
  __syncthreads();

  int lenv[8];
#pragma unroll
  for (int r = 0; r < 8; r++) lenv[r] = len_s[r + 8 * khalf];

  int   n0s[8];
  float biasv[8];
#pragma unroll
  for (int tt = 0; tt < 8; tt++) {
    n0s[tt]   = (tt >> 1) * H_ + wave * 32 + (tt & 1) * 16;
    biasv[tt] = bias[n0s[tt] + nlane];
  }

  v8f hreg[2], creg[2];
#pragma unroll
  for (int half = 0; half < 2; half++)
#pragma unroll
    for (int r = 0; r < 8; r++) { hreg[half][r] = 0.f; creg[half][r] = 0.f; }

  // ---- build initial x tile (step 0) into buffer 0 ----
  {
    const int tp0 = dir ? (T_ - 1) : 0;
    const int tok = tok_lds[wave][tp0];
    const float4* src = (const float4*)(emb + (size_t)tok * E_ + lane * 16);
    unsigned pk[8];
#pragma unroll
    for (int q = 0; q < 4; q++) {
      float4 v = src[q];
      pk[q * 2 + 0] = (unsigned)f2bf(v.x) | ((unsigned)f2bf(v.y) << 16);
      pk[q * 2 + 1] = (unsigned)f2bf(v.z) | ((unsigned)f2bf(v.w) << 16);
    }
    uint4* dst = (uint4*)&x_lds[0][wave][lane * 16];
    dst[0] = make_uint4(pk[0], pk[1], pk[2], pk[3]);
    dst[1] = make_uint4(pk[4], pk[5], pk[6], pk[7]);
  }
  __syncthreads();

  const unsigned short* hrow = &h_lds[m][0];
  int cur = 0;

#pragma unroll 1
  for (int s = 0; s < T_; s++) {
    const int  tp        = dir ? (T_ - 1 - s) : s;
    const bool have_next = (s + 1 < T_);

    // ---- issue next step's embedding gather; lands during the matmul ----
    float4 xv[4];
    if (have_next) {
      const int tpn = dir ? (T_ - 2 - s) : (s + 1);
      const int tok = tok_lds[wave][tpn];
      const float4* src = (const float4*)(emb + (size_t)tok * E_ + lane * 16);
      xv[0] = src[0]; xv[1] = src[1]; xv[2] = src[2]; xv[3] = src[3];
    }

    // ---- gates = bias + x@W_ih^T + h@W_hh^T  (bf16 WMMA, f32 accum) ----
    const unsigned short* xrow = &x_lds[cur][m][0];
    v8f acc[8];
#pragma unroll
    for (int tt = 0; tt < 8; tt++)
#pragma unroll
      for (int e = 0; e < 8; e++) acc[tt][e] = biasv[tt];

#pragma unroll 1
    for (int k0 = 0; k0 < E_; k0 += 32) {
      v16bf a = load_a(xrow, k0, khalf);
      __builtin_prefetch(wih + (size_t)(k0 + 32 + lane) * G_ + n0s[0], 0, 1);
#pragma unroll
      for (int tt = 0; tt < 8; tt++) {
        v16bf b = load_b(wih, k0, n0s[tt]);
        acc[tt] = __builtin_amdgcn_wmma_f32_16x16x32_bf16(
            false, a, false, b, (short)0, acc[tt], false, false);
      }
    }
#pragma unroll 1
    for (int k0 = 0; k0 < H_; k0 += 32) {
      v16bf a = load_a(hrow, k0, khalf);
      __builtin_prefetch(whh + (size_t)(k0 + 32 + lane) * G_ + n0s[0], 0, 1);
#pragma unroll
      for (int tt = 0; tt < 8; tt++) {
        v16bf b = load_b(whh, k0, n0s[tt]);
        acc[tt] = __builtin_amdgcn_wmma_f32_16x16x32_bf16(
            false, a, false, b, (short)0, acc[tt], false, false);
      }
    }
    __syncthreads();   // all x/h LDS reads for step s complete

    // ---- elementwise LSTM cell + variable-length masking ----
#pragma unroll
    for (int half = 0; half < 2; half++) {
      const int jc = wave * 32 + half * 16 + nlane;
#pragma unroll
      for (int r = 0; r < 8; r++) {
        float ig = sigm(acc[0 + half][r]);
        float fg = sigm(acc[2 + half][r]);
        float gg = tanh_f(acc[4 + half][r]);
        float og = sigm(acc[6 + half][r]);
        float c2 = fg * creg[half][r] + ig * gg;
        float h2 = og * tanh_f(c2);
        bool valid = tp < lenv[r];
        float cn = valid ? c2 : creg[half][r];
        float hn = valid ? h2 : hreg[half][r];
        creg[half][r] = cn;
        hreg[half][r] = hn;
        h_lds[r + 8 * khalf][jc] = f2bf(hn);
      }
    }

    // ---- commit prefetched x tile into the other buffer ----
    if (have_next) {
      unsigned pk[8];
#pragma unroll
      for (int q = 0; q < 4; q++) {
        float4 v = xv[q];
        pk[q * 2 + 0] = (unsigned)f2bf(v.x) | ((unsigned)f2bf(v.y) << 16);
        pk[q * 2 + 1] = (unsigned)f2bf(v.z) | ((unsigned)f2bf(v.w) << 16);
      }
      uint4* dst = (uint4*)&x_lds[cur ^ 1][wave][lane * 16];
      dst[0] = make_uint4(pk[0], pk[1], pk[2], pk[3]);
      dst[1] = make_uint4(pk[4], pk[5], pk[6], pk[7]);
    }
    __syncthreads();   // h/x LDS updates visible before next step's matmul
    cur ^= 1;
  }

  // ---- emit final hidden state: hidden[b][dir*512 + j], f32 ----
#pragma unroll
  for (int half = 0; half < 2; half++) {
    const int jc = wave * 32 + half * 16 + nlane;
#pragma unroll
    for (int r = 0; r < 8; r++) {
      const int b = b0 + r + 8 * khalf;
      hidden[(size_t)b * (2 * H_) + dir * H_ + jc] = hreg[half][r];
    }
  }
}

// ---------------------------------------------------------------------------
// Tail GEMVs (tiny: 67M + 0.7M MACs) — plain f32.
// ---------------------------------------------------------------------------
__global__ void k_transform(const float* __restrict__ hidden,
                            const float* __restrict__ tw, float* __restrict__ trans) {
  int idx = blockIdx.x * blockDim.x + threadIdx.x;   // B_*H_ outputs
  if (idx >= B_ * H_) return;
  int b = idx >> 9, j = idx & 511;
  const float* hr = hidden + (size_t)b * (2 * H_);
  const float* wr = tw + (size_t)j * (2 * H_);
  float acc = 0.f;
  for (int k = 0; k < 2 * H_; k++) acc = fmaf(hr[k], wr[k], acc);
  trans[idx] = acc;
}

__global__ void k_fc(const float* __restrict__ trans, const float* __restrict__ fw,
                     const float* __restrict__ fb, float* __restrict__ out) {
  int idx = blockIdx.x * blockDim.x + threadIdx.x;   // B_*C_ outputs
  if (idx >= B_ * C_) return;
  int b = idx / C_, c = idx % C_;
  const float* tr = trans + (size_t)b * H_;
  const float* wr = fw + (size_t)c * H_;
  float acc = fb[c];
  for (int k = 0; k < H_; k++) acc = fmaf(tr[k], wr[k], acc);
  out[idx] = acc;
}

// ---------------------------------------------------------------------------
extern "C" void kernel_launch(void* const* d_in, const int* in_sizes, int n_in,
                              void* d_out, int out_size, void* d_ws, size_t ws_size,
                              hipStream_t stream) {
  (void)in_sizes; (void)n_in; (void)out_size; (void)ws_size;

  const int*   text        = (const int*)  d_in[0];
  const float* emb         = (const float*)d_in[1];
  const float* w_ih_f      = (const float*)d_in[2];
  const float* w_hh_f      = (const float*)d_in[3];
  const float* b_ih_f      = (const float*)d_in[4];
  const float* b_hh_f      = (const float*)d_in[5];
  const float* w_ih_r      = (const float*)d_in[6];
  const float* w_hh_r      = (const float*)d_in[7];
  const float* b_ih_r      = (const float*)d_in[8];
  const float* b_hh_r      = (const float*)d_in[9];
  const float* transform_w = (const float*)d_in[10];
  const float* fc_w        = (const float*)d_in[11];
  const float* fc_b        = (const float*)d_in[12];
  float* out = (float*)d_out;

  // Workspace layout (~8.77 MiB total)
  char* ws = (char*)d_ws;
  const size_t W = (size_t)512 * G_ * sizeof(unsigned short);   // 2 MB per weight
  unsigned short* wT_ih_f = (unsigned short*)(ws + 0 * W);
  unsigned short* wT_hh_f = (unsigned short*)(ws + 1 * W);
  unsigned short* wT_ih_r = (unsigned short*)(ws + 2 * W);
  unsigned short* wT_hh_r = (unsigned short*)(ws + 3 * W);
  float* bias_f = (float*)(ws + 4 * W);
  float* bias_r = bias_f + G_;
  int*   lens   = (int*)(bias_r + G_);
  float* hidden = (float*)(ws + 4 * W + (size_t)20 * 1024);         // [128][1024]
  float* trans  = hidden + (size_t)B_ * 2 * H_;                     // [128][512]

  const int nW = G_ * 512;
  k_wT_bf16 <<<(nW + 255) / 256, 256, 0, stream>>>(w_ih_f, wT_ih_f);
  k_wT_bf16 <<<(nW + 255) / 256, 256, 0, stream>>>(w_hh_f, wT_hh_f);
  k_wT_bf16 <<<(nW + 255) / 256, 256, 0, stream>>>(w_ih_r, wT_ih_r);
  k_wT_bf16 <<<(nW + 255) / 256, 256, 0, stream>>>(w_hh_r, wT_hh_r);
  k_bias    <<<(G_ + 255) / 256, 256, 0, stream>>>(b_ih_f, b_hh_f, bias_f);
  k_bias    <<<(G_ + 255) / 256, 256, 0, stream>>>(b_ih_r, b_hh_r, bias_r);
  k_lengths <<<B_, T_, 0, stream>>>(text, lens);

  k_bilstm  <<<16, 512, 0, stream>>>(text, emb, wT_ih_f, wT_hh_f, wT_ih_r, wT_hh_r,
                                     bias_f, bias_r, lens, hidden);

  k_transform <<<(B_ * H_ + 255) / 256, 256, 0, stream>>>(hidden, transform_w, trans);
  k_fc        <<<(B_ * C_ + 255) / 256, 256, 0, stream>>>(trans, fc_w, fc_b, out);
}
